// TropicalCausalSelfAttention_24292335026390
// MI455X (gfx1250) — compile-verified, matrix operand order checked by reading, not run on hardware
//
#include <hip/hip_runtime.h>
#include <hip/hip_bf16.h>

typedef __attribute__((ext_vector_type(16))) _Float16 v16h;
typedef __attribute__((ext_vector_type(8)))  _Float16 v8h;
typedef __attribute__((ext_vector_type(8)))  float    v8f;

#define TT  512   // sequence length
#define CC  512   // model dim
#define HH  8     // query heads
#define KVH 4     // kv heads
#define DH  64    // head dim

// ---------------------------------------------------------------------------
// CDNA5 async global->LDS copy (16 bytes / lane) + async counter wait.
// ---------------------------------------------------------------------------
__device__ __forceinline__ void async_copy_b128(unsigned lds_off,
                                                const void* gsrc) {
  asm volatile("global_load_async_to_lds_b128 %0, %1, off"
               :: "v"(lds_off), "v"(gsrc) : "memory");
}
__device__ __forceinline__ void wait_async0() {
  asm volatile("s_wait_asynccnt 0x0" ::: "memory");
}

// ---------------------------------------------------------------------------
// Kernel 1: fp32 -> f16 conversion of activations and weights.
// wh = concat(w_q (512x512), w_k (256x512), w_v (256x512)) row-major, K=512.
// ---------------------------------------------------------------------------
__global__ __launch_bounds__(256) void cvt_f16_kernel(
    const float* __restrict__ x,  const float* __restrict__ wq,
    const float* __restrict__ wk, const float* __restrict__ wv,
    const float* __restrict__ wp,
    _Float16* __restrict__ xh, _Float16* __restrict__ wh,
    _Float16* __restrict__ wph) {
  const int i = blockIdx.x * blockDim.x + threadIdx.x;   // 0 .. 524287
  if (i < 262144) {
    xh[i]  = (_Float16)x[i];
    wph[i] = (_Float16)wp[i];
    wh[i]  = (_Float16)wq[i];
  } else if (i < 393216) {
    wh[i] = (_Float16)wk[i - 262144];
  } else if (i < 524288) {
    wh[i] = (_Float16)wv[i - 393216];
  }
}

// ---------------------------------------------------------------------------
// Kernel 2/5: C(MxN,f32) = A(MxK,f16 row-major) * W(NxK,f16 row-major)^T
// Block = 128 threads (4 waves), 64x64 output tile, double-buffered LDS:
//   - B slab (64 N-rows x 32 k = 4 KB) for step i+1 staged asynchronously
//     (global_load_async_to_lds_b128, 2 chunks/thread) while step i computes.
//   - Single barrier per k-step: proves RAW on buf_i (after own
//     s_wait_asynccnt 0) and WAR on buf_{i+1} (last read in step i-1).
//   - All 4 B fragments loaded from LDS first (one ds clause), then 4
//     back-to-back WMMAs reusing one A fragment.
// Fragment layout per CDNA5 ISA 7.12.2:
//   A: lane l -> M = l&15 ; element e -> K = (e<8?e:e+8) + 8*(l>>4)
//   B: lane l -> N = l&15 ; same K mapping (W rows are the K-major vectors)
//   C: vgpr r -> M = r + 8*(l>>4) ; N = l&15
// ---------------------------------------------------------------------------
__global__ __launch_bounds__(128) void gemm_f16_wmma_kernel(
    const _Float16* __restrict__ A, const _Float16* __restrict__ W,
    float* __restrict__ C, int M, int N, int K) {
  __shared__ _Float16 Bs[2][64 * 32];        // 2 x 4 KB slabs

  const int tid  = threadIdx.x;
  const int lane = tid & 31;
  const int wv   = tid >> 5;                 // wave 0..3
  const int nblk = N >> 6;
  const int bM   = blockIdx.x / nblk;
  const int bN   = blockIdx.x - bM * nblk;
  const int mbase = bM * 64 + wv * 16;
  const int nbase = bN * 64;

  const int hs = lane >> 4;                  // K-half select
  const int mr = lane & 15;
  const _Float16* arow = A + (size_t)(mbase + mr) * K + hs * 8;

  // async staging map: 256 16-byte chunks (64 rows x 4), 2 per thread
  const int c0 = tid * 2;
  const int r0 = c0 >> 2, q0 = c0 & 3;
  const int c1 = c0 + 1;
  const int r1 = c1 >> 2, q1 = c1 & 3;
  const _Float16* g0 = W + (size_t)(nbase + r0) * K + q0 * 8;
  const _Float16* g1 = W + (size_t)(nbase + r1) * K + q1 * 8;
  const unsigned lbase = (unsigned)(size_t)(void*)&Bs[0][0];
  const unsigned o0 = (unsigned)(r0 * 64 + q0 * 16);
  const unsigned o1 = (unsigned)(r1 * 64 + q1 * 16);

  // prologue: stage k-step 0 into buffer 0
  async_copy_b128(lbase + o0, (const void*)g0);
  async_copy_b128(lbase + o1, (const void*)g1);

  v8f acc[4] = {{}, {}, {}, {}};
  const int nsteps = K >> 5;

  for (int i = 0; i < nsteps; ++i) {
    const int kk = i << 5;
    wait_async0();                           // buf_i copies (mine) landed
    __syncthreads();                         // all waves: buf_i visible,
                                             // buf_{i+1} readers done (i-1)
    if (i + 1 < nsteps) {                    // pipeline next slab
      const unsigned nb = lbase + (unsigned)(((i + 1) & 1) * 4096);
      async_copy_b128(nb + o0, (const void*)(g0 + kk + 32));
      async_copy_b128(nb + o1, (const void*)(g1 + kk + 32));
      __builtin_prefetch(arow + kk + 32, 0, 0);
    }

    // A fragment from global (wave-private rows)
    v8h a0 = *(const v8h*)(arow + kk);
    v8h a1 = *(const v8h*)(arow + kk + 16);
    v16h a;
#pragma unroll
    for (int e = 0; e < 8; ++e) { a[e] = a0[e]; a[e + 8] = a1[e]; }

    // all 4 B fragments from LDS (one ds_load_b128 clause), then 4 WMMAs
    const _Float16* bsl = &Bs[i & 1][0];
    v16h b[4];
#pragma unroll
    for (int nt = 0; nt < 4; ++nt) {
      const _Float16* bl = bsl + (nt * 16 + mr) * 32 + hs * 8;
      v8h b0 = *(const v8h*)(bl);
      v8h b1 = *(const v8h*)(bl + 16);
#pragma unroll
      for (int e = 0; e < 8; ++e) { b[nt][e] = b0[e]; b[nt][e + 8] = b1[e]; }
    }
#pragma unroll
    for (int nt = 0; nt < 4; ++nt)
      acc[nt] = __builtin_amdgcn_wmma_f32_16x16x32_f16(
          false, a, false, b[nt], (short)0, acc[nt], false, false);
  }

  const int row0 = mbase + hs * 8;
#pragma unroll
  for (int nt = 0; nt < 4; ++nt) {
    const int col = nbase + nt * 16 + mr;
#pragma unroll
    for (int r = 0; r < 8; ++r)
      C[(size_t)(row0 + r) * N + col] = acc[nt][r];
  }
}

// ---------------------------------------------------------------------------
// Kernel 3: rope + rms_norm + center for q/k, center for v.
// grid = T * 16 blocks of 64 threads. unit<8: q head; 8..11: k head; 12..15: v.
//   qn[h][t][d], kn[kvh][t][d], vn[kvh][t][d]
// ---------------------------------------------------------------------------
__global__ __launch_bounds__(64) void rope_rms_center_kernel(
    const float* __restrict__ qkv, const float* __restrict__ cosp,
    const float* __restrict__ sinp, float* __restrict__ qn,
    float* __restrict__ kn, float* __restrict__ vn) {
  __shared__ float red[64];
  const int t    = blockIdx.x >> 4;
  const int unit = blockIdx.x & 15;
  const int d    = threadIdx.x;

  const float* row;
  if (unit < 8)       row = qkv + (size_t)t * 1024 + unit * 64;
  else if (unit < 12) row = qkv + (size_t)t * 1024 + 512 + (unit - 8) * 64;
  else                row = qkv + (size_t)t * 1024 + 768 + (unit - 12) * 64;

  float val;
  if (unit < 12) {                           // rope + rms (q and k)
    const int dd = d & 31;
    const float c  = cosp[t * 32 + dd];
    const float s  = sinp[t * 32 + dd];
    const float x1 = row[dd];
    const float x2 = row[dd + 32];
    val = (d < 32) ? (x1 * c + x2 * s) : (x2 * c - x1 * s);
    red[d] = val * val;
    __syncthreads();
    for (int off = 32; off >= 1; off >>= 1) {
      if (d < off) red[d] += red[d + off];
      __syncthreads();
    }
    val *= rsqrtf(red[0] * (1.0f / 64.0f) + 1e-6f);
    __syncthreads();                          // red reused below
  } else {
    val = row[d];
  }

  // center over d (subtract max)
  red[d] = val;
  __syncthreads();
  for (int off = 32; off >= 1; off >>= 1) {
    if (d < off) red[d] = fmaxf(red[d], red[d + off]);
    __syncthreads();
  }
  val -= red[0];

  if (unit < 8)       qn[((size_t)unit * TT + t) * DH + d] = val;
  else if (unit < 12) kn[((size_t)(unit - 8) * TT + t) * DH + d] = val;
  else                vn[((size_t)(unit - 12) * TT + t) * DH + d] = val;
}

// ---------------------------------------------------------------------------
// Kernel 4: tropical attention. Block per (h,i).
//   sc[j] = max_d(q[i,d] + k[j,d]) for j<=i (causal)
//   y[d]  = max_j(sc[j] + v[j,d]);  output centered over d (score row-max
//   subtraction cancels against the centering, so it is omitted).
// Result stored as f16 in y-layout [t][h*64+d] for the output GEMM.
// ---------------------------------------------------------------------------
__global__ __launch_bounds__(256) void tropical_attn_kernel(
    const float* __restrict__ qn, const float* __restrict__ kn,
    const float* __restrict__ vn, _Float16* __restrict__ yh) {
  __shared__ float qs[64];
  __shared__ float sc[TT];
  __shared__ float red[64];
  const int tid = threadIdx.x;
  const int h   = blockIdx.x >> 9;   // / T
  const int i   = blockIdx.x & (TT - 1);
  const int kvh = h >> 1;            // GQA repeat factor 2

  if (tid < 64) qs[tid] = qn[((size_t)h * TT + i) * DH + tid];
  __syncthreads();

  const float* kbase = kn + (size_t)kvh * TT * DH;
  for (int j = tid; j <= i; j += 256) {
    const float* kr = kbase + (size_t)j * DH;
    float m = -INFINITY;
#pragma unroll
    for (int d = 0; d < 64; ++d) m = fmaxf(m, qs[d] + kr[d]);
    sc[j] = m;
  }
  __syncthreads();

  float y = -INFINITY;
  if (tid < 64) {
    const float* vcol = vn + (size_t)kvh * TT * DH + tid;
    for (int j = 0; j <= i; ++j)
      y = fmaxf(y, sc[j] + vcol[(size_t)j * DH]);
    red[tid] = y;
  }
  __syncthreads();
  for (int off = 32; off >= 1; off >>= 1) {
    if (tid < off) red[tid] = fmaxf(red[tid], red[tid + off]);
    __syncthreads();
  }
  if (tid < 64)
    yh[(size_t)i * (HH * DH) + h * DH + tid] = (_Float16)(y - red[0]);
}

// ---------------------------------------------------------------------------
// Kernel 6: final centering of the projected output (row max over 512).
// ---------------------------------------------------------------------------
__global__ __launch_bounds__(256) void center512_kernel(
    const float* __restrict__ yo, float* __restrict__ out) {
  __shared__ float red[256];
  const int t = blockIdx.x, tid = threadIdx.x;
  const float* row = yo + (size_t)t * 512;
  const float a = row[tid];
  const float b = row[tid + 256];
  red[tid] = fmaxf(a, b);
  __syncthreads();
  for (int off = 128; off >= 1; off >>= 1) {
    if (tid < off) red[tid] = fmaxf(red[tid], red[tid + off]);
    __syncthreads();
  }
  const float mx = red[0];
  out[(size_t)t * 512 + tid]       = a - mx;
  out[(size_t)t * 512 + tid + 256] = b - mx;
}

// ---------------------------------------------------------------------------
extern "C" void kernel_launch(void* const* d_in, const int* in_sizes, int n_in,
                              void* d_out, int out_size, void* d_ws,
                              size_t ws_size, hipStream_t stream) {
  const float* x    = (const float*)d_in[0];
  const float* cosp = (const float*)d_in[1];
  const float* sinp = (const float*)d_in[2];
  const float* wq   = (const float*)d_in[3];
  const float* wk   = (const float*)d_in[4];
  const float* wv   = (const float*)d_in[5];
  const float* wp   = (const float*)d_in[6];
  float* out = (float*)d_out;

  char* ws = (char*)d_ws;
  _Float16* xh  = (_Float16*)(ws + 0);          //  512*512 f16
  _Float16* wh  = (_Float16*)(ws + 524288);     // 1024*512 f16
  _Float16* wph = (_Float16*)(ws + 1572864);    //  512*512 f16
  float*    qkv = (float*)  (ws + 2097152);     //  512*1024 f32
  float*    qn  = (float*)  (ws + 4194304);     //  8*512*64 f32
  float*    kn  = (float*)  (ws + 5242880);     //  4*512*64 f32
  float*    vn  = (float*)  (ws + 5767168);     //  4*512*64 f32
  _Float16* yh  = (_Float16*)(ws + 6291456);    //  512*512 f16
  float*    yo  = (float*)  (ws + 6815744);     //  512*512 f32

  // 1. convert activations + weights to f16
  cvt_f16_kernel<<<2048, 256, 0, stream>>>(x, wq, wk, wv, wp, xh, wh, wph);
  // 2. fused QKV projection: 64x64 block tiles -> 8*16 = 128 blocks
  gemm_f16_wmma_kernel<<<128, 128, 0, stream>>>(xh, wh, qkv, TT, 1024, CC);
  // 3. rope + rms + center
  rope_rms_center_kernel<<<TT * 16, 64, 0, stream>>>(qkv, cosp, sinp, qn, kn, vn);
  // 4. tropical (max-plus) causal attention
  tropical_attn_kernel<<<HH * TT, 256, 0, stream>>>(qn, kn, vn, yh);
  // 5. output projection: 64x64 block tiles -> 8*8 = 64 blocks
  gemm_f16_wmma_kernel<<<64, 128, 0, stream>>>(yh, wph, yo, TT, CC, HH * DH);
  // 6. final row centering
  center512_kernel<<<TT, 256, 0, stream>>>(yo, out);
}
